// HPCrnn_72928544686729
// MI455X (gfx1250) — matrix operand order repaired and checked by reading, not compiled, not run on hardware
//
#include <hip/hip_runtime.h>
#include <hip/hip_bf16.h>
#include <stdint.h>

// Problem constants (all dims 1024 per the reference setup)
#define NN 1024
#define TLEN 100
#define CUE_STEP 10

typedef __bf16 bf16;
typedef __attribute__((ext_vector_type(8)))  __bf16 v8bf;
typedef __attribute__((ext_vector_type(16))) __bf16 v16bf;
typedef __attribute__((ext_vector_type(8)))  float  v8f;

// ---- output layout (floats), concatenated in reference return order ----
// act_cell(1024*2), ec3his(100*1024), ec5his(100*1024), ca1his(100*1024),
// ec3(1024^2), ec5(1024^2), ca1(1024^2)
#define OFF_ACT 0
#define OFF_E3H 2048
#define OFF_E5H (2048 + 102400)
#define OFF_C1H (2048 + 2*102400)
#define OFF_EC3 (2048 + 3*102400)
#define OFF_EC5 (OFF_EC3 + NN*NN)
#define OFF_CA1 (OFF_EC5 + NN*NN)

__device__ __forceinline__ v16bf cat16(v8bf lo, v8bf hi) {
  return __builtin_shufflevector(lo, hi, 0,1,2,3,4,5,6,7,8,9,10,11,12,13,14,15);
}

__device__ __forceinline__ float sigmoidf_(float x) {
  return 1.0f / (1.0f + __expf(-x));
}

// JAX-compatible threefry2x32 (20 rounds, key schedule per jax._src.prng)
__device__ __forceinline__ void threefry2x32(uint32_t k0, uint32_t k1,
                                             uint32_t c0, uint32_t c1,
                                             uint32_t& o0, uint32_t& o1) {
  uint32_t ks0 = k0, ks1 = k1, ks2 = 0x1BD11BDAu ^ k0 ^ k1;
  uint32_t x0 = c0 + ks0;
  uint32_t x1 = c1 + ks1;
  const int rotA[4] = {13, 15, 26, 6};
  const int rotB[4] = {17, 29, 16, 24};
  uint32_t ks[3] = {ks0, ks1, ks2};
#pragma unroll
  for (int i = 0; i < 5; ++i) {
    const int* rot = (i & 1) ? rotB : rotA;
#pragma unroll
    for (int r = 0; r < 4; ++r) {
      x0 += x1;
      x1 = (x1 << rot[r]) | (x1 >> (32 - rot[r]));
      x1 ^= x0;
    }
    x0 += ks[(i + 1) % 3];
    x1 += ks[(i + 2) % 3] + (uint32_t)(i + 1);
  }
  o0 = x0; o1 = x1;
}

// ------------------------- setup kernels -------------------------

// Convert ec3/ec5 state to workspace; convert+transpose both weight matrices
// to bf16, stored N-major so WMMA B fragments are contiguous per lane.
__global__ void __launch_bounds__(256) k_prep(
    const float* __restrict__ ec3_last, const float* __restrict__ ec5_last,
    const float* __restrict__ w1, const float* __restrict__ w2,
    float* __restrict__ ec3f, float* __restrict__ ec5f,
    bf16* __restrict__ ec3b, bf16* __restrict__ wt1, bf16* __restrict__ wt2) {
  int i = blockIdx.x * blockDim.x + threadIdx.x;
  if (i >= NN * NN) return;
  float e3 = ec3_last[i];
  ec3f[i] = e3;
  ec3b[i] = (bf16)e3;
  ec5f[i] = ec5_last[i];
  int n = i / NN, k = i % NN;          // wt[n][k] = w[k][n]
  wt1[i] = (bf16)w1[k * NN + n];
  wt2[i] = (bf16)w2[k * NN + n];
}

// noise_keys = jax.random.split(key(42), 100): key t = out_flat[2t], out_flat[2t+1]
__global__ void k_keys(uint32_t* __restrict__ keys) {
  int t = threadIdx.x;
  if (t >= TLEN) return;
  const uint32_t K0 = 0u, K1 = 42u;
  uint32_t o0, o1, r0, r1;
  if (t < 50) {  // both flat indices < 100 -> word0 of pairs (j, j+100)
    threefry2x32(K0, K1, (uint32_t)(2 * t),     (uint32_t)(2 * t + 100), o0, o1); r0 = o0;
    threefry2x32(K0, K1, (uint32_t)(2 * t + 1), (uint32_t)(2 * t + 101), o0, o1); r1 = o0;
  } else {       // flat indices >= 100 -> word1 of pairs (j-100, j)
    threefry2x32(K0, K1, (uint32_t)(2 * t - 100), (uint32_t)(2 * t),     o0, o1); r0 = o1;
    threefry2x32(K0, K1, (uint32_t)(2 * t - 99),  (uint32_t)(2 * t + 1), o0, o1); r1 = o1;
  }
  keys[2 * t] = r0;
  keys[2 * t + 1] = r1;
}

__global__ void __launch_bounds__(256) k_ca3(float* __restrict__ ca3) {
  int i = blockIdx.x * blockDim.x + threadIdx.x;
  if (i >= TLEN * NN) return;
  int t = i / NN, c = i % NN;
  float d = (100.0f * (float)c) / 1023.0f - (float)t;  // linspace(0,100,1024)[c] - x
  ca3[i] = __expf(-d * d * 0.02f);                      // 1/(2*sigma^2) = 0.02
}

__global__ void __launch_bounds__(256) k_drive(
    const float* __restrict__ ca3, const float* __restrict__ w,
    float* __restrict__ drive) {
  int i = blockIdx.x * blockDim.x + threadIdx.x;
  if (i >= TLEN * NN) return;
  int t = i / NN, j = i % NN;
  float acc = 0.0f;
  for (int c = 0; c < NN; ++c) acc = fmaf(ca3[t * NN + c], w[c * NN + j], acc);
  drive[i] = acc;
}

// ------------------------- WMMA GEMM core -------------------------
// One wave computes a 32x32 output tile: 2 A fragments x 2 B fragments feed
// 4 WMMAs per K-chunk (8 b128 loads / 4 WMMA = 2 loads per WMMA, each
// fragment reused twice). A: row-major MxK bf16. Bt: weight transposed
// (N-major, N x K). Fragment maps follow the CDNA5 ISA VGPR layout tables.
__device__ __forceinline__ void wmma_tile32(
    const bf16* __restrict__ A, const bf16* __restrict__ Bt,
    int m_base, int n_base, int half, int l16,
    v8f& acc00, v8f& acc01, v8f& acc10, v8f& acc11) {
  const bf16* aRow0 = A  + (m_base + l16) * NN;
  const bf16* aRow1 = A  + (m_base + 16 + l16) * NN;
  const bf16* bRow0 = Bt + (n_base + l16) * NN;
  const bf16* bRow1 = Bt + (n_base + 16 + l16) * NN;
  for (int k = 0; k < NN; k += 32) {
    if (k + 256 < NN) {
      __builtin_prefetch(aRow0 + k + 256, 0, 1);  // global_prefetch_b8
      __builtin_prefetch(bRow0 + k + 256, 0, 1);
    }
    // A 16x32 bf16: lanes<16 hold K {0..7,16..23}, lanes>=16 hold {8..15,24..31}
    v16bf a0 = cat16(*(const v8bf*)(aRow0 + k + half * 8),
                     *(const v8bf*)(aRow0 + k + 16 + half * 8));
    v16bf a1 = cat16(*(const v8bf*)(aRow1 + k + half * 8),
                     *(const v8bf*)(aRow1 + k + 16 + half * 8));
    // B 32x16 bf16: lane holds column n, K {0..7,16..23} or {8..15,24..31}
    const bf16* b0p = bRow0 + k + half * 8;
    const bf16* b1p = bRow1 + k + half * 8;
    v16bf b0 = cat16(*(const v8bf*)b0p, *(const v8bf*)(b0p + 16));
    v16bf b1 = cat16(*(const v8bf*)b1p, *(const v8bf*)(b1p + 16));
    acc00 = __builtin_amdgcn_wmma_f32_16x16x32_bf16(false, a0, false, b0,
                                                    (short)0, acc00, false, false);
    acc01 = __builtin_amdgcn_wmma_f32_16x16x32_bf16(false, a0, false, b1,
                                                    (short)0, acc01, false, false);
    acc10 = __builtin_amdgcn_wmma_f32_16x16x32_bf16(false, a1, false, b0,
                                                    (short)0, acc10, false, false);
    acc11 = __builtin_amdgcn_wmma_f32_16x16x32_bf16(false, a1, false, b1,
                                                    (short)0, acc11, false, false);
  }
}

// Step part 1: S = ec3 @ wec3ca1 ; ca1 = relu(drive * (1 + sigmoid(S)) - bias)
// Grid: (8, 16), block 256 = 8 waves arranged 2 (m) x 4 (n); WG tile 64x128.
__global__ void __launch_bounds__(256) k_step1(
    const bf16* __restrict__ ec3b, const bf16* __restrict__ wt1,
    const float* __restrict__ drive_t, const float* __restrict__ bias,
    float* __restrict__ ca1f, bf16* __restrict__ ca1b,
    float* __restrict__ ca1his_t) {
  int wave = threadIdx.x >> 5, lane = threadIdx.x & 31;
  int half = lane >> 4, l16 = lane & 15;
  int m_base = blockIdx.y * 64  + (wave >> 2) * 32;
  int n_base = blockIdx.x * 128 + (wave & 3) * 32;
  v8f acc00 = {}, acc01 = {}, acc10 = {}, acc11 = {};
  wmma_tile32(ec3b, wt1, m_base, n_base, half, l16, acc00, acc01, acc10, acc11);

  int n0 = n_base + l16, n1 = n0 + 16;
  float d0 = drive_t[n0], d1 = drive_t[n1];
  float bb0 = bias[n0],   bb1 = bias[n1];
#pragma unroll
  for (int r = 0; r < 8; ++r) {
    int m0 = m_base + half * 8 + r;  // C/D layout: lanes>=16 hold rows M=8..15
    int m1 = m0 + 16;
    float c00 = fmaxf(d0 * (1.0f + sigmoidf_(acc00[r])) - bb0, 0.0f);
    float c01 = fmaxf(d1 * (1.0f + sigmoidf_(acc01[r])) - bb1, 0.0f);
    float c10 = fmaxf(d0 * (1.0f + sigmoidf_(acc10[r])) - bb0, 0.0f);
    float c11 = fmaxf(d1 * (1.0f + sigmoidf_(acc11[r])) - bb1, 0.0f);
    ca1f[m0 * NN + n0] = c00;  ca1b[m0 * NN + n0] = (bf16)c00;
    ca1f[m0 * NN + n1] = c01;  ca1b[m0 * NN + n1] = (bf16)c01;
    ca1f[m1 * NN + n0] = c10;  ca1b[m1 * NN + n0] = (bf16)c10;
    ca1f[m1 * NN + n1] = c11;  ca1b[m1 * NN + n1] = (bf16)c11;
    if (m0 == 0) { ca1his_t[n0] = c00; ca1his_t[n1] = c01; }
  }
}

// Elementwise tail of step part 2 for one output element.
__device__ __forceinline__ void step2_elem(
    int m, int n, float p, int step,
    const uint8_t* __restrict__ cue, uint32_t k0, uint32_t k1,
    float* __restrict__ ec3f, float* __restrict__ ec5f,
    bf16* __restrict__ ec3b,
    float* __restrict__ ec3his_t, float* __restrict__ ec5his_t) {
  int idx = m * NN + n;                               // 10*TS == 1.0
  float e5 = 0.69f + 0.3f * sigmoidf_(4.0f * (ec5f[idx] + p - 0.3f));
  float e3 = e5 * ec3f[idx];
  if (step == CUE_STEP && cue[idx]) e3 = 0.4f * e3 + 0.6f;
  // jax.random.uniform(nkey, (1024,1024)) < NOISE_P via threefry2x32
  uint32_t i = (uint32_t)idx;
  const uint32_t H = 1u << 19;  // half of 1024*1024 elements (JAX uniform split)
  uint32_t c0 = (i < H) ? i : i - H;
  uint32_t c1 = (i < H) ? i + H : i;
  uint32_t o0, o1;
  threefry2x32(k0, k1, c0, c1, o0, o1);
  uint32_t bits = (i < H) ? o0 : o1;
  float u = __uint_as_float((bits >> 9) | 0x3F800000u) - 1.0f;
  if (u < 0.004f) e3 = 0.5f * e3 + 0.3f;
  ec5f[idx] = e5;
  ec3f[idx] = e3;
  ec3b[idx] = (bf16)e3;
  if (m == 0) { ec3his_t[n] = e3; ec5his_t[n] = e5; }
}

// Step part 2: P = ca1 @ wca1ec5 ; ec5 = g(ec5 + P) ; ec3 = ec5*ec3 + cue + noise
__global__ void __launch_bounds__(256) k_step2(
    const bf16* __restrict__ ca1b, const bf16* __restrict__ wt2,
    const uint8_t* __restrict__ cue, const uint32_t* __restrict__ keys, int step,
    float* __restrict__ ec3f, float* __restrict__ ec5f, bf16* __restrict__ ec3b,
    float* __restrict__ ec3his_t, float* __restrict__ ec5his_t) {
  int wave = threadIdx.x >> 5, lane = threadIdx.x & 31;
  int half = lane >> 4, l16 = lane & 15;
  int m_base = blockIdx.y * 64  + (wave >> 2) * 32;
  int n_base = blockIdx.x * 128 + (wave & 3) * 32;
  v8f acc00 = {}, acc01 = {}, acc10 = {}, acc11 = {};
  wmma_tile32(ca1b, wt2, m_base, n_base, half, l16, acc00, acc01, acc10, acc11);

  uint32_t k0 = keys[2 * step], k1 = keys[2 * step + 1];
  int n0 = n_base + l16, n1 = n0 + 16;
#pragma unroll
  for (int r = 0; r < 8; ++r) {
    int m0 = m_base + half * 8 + r;
    int m1 = m0 + 16;
    step2_elem(m0, n0, acc00[r], step, cue, k0, k1, ec3f, ec5f, ec3b, ec3his_t, ec5his_t);
    step2_elem(m0, n1, acc01[r], step, cue, k0, k1, ec3f, ec5f, ec3b, ec3his_t, ec5his_t);
    step2_elem(m1, n0, acc10[r], step, cue, k0, k1, ec3f, ec5f, ec3b, ec3his_t, ec5his_t);
    step2_elem(m1, n1, acc11[r], step, cue, k0, k1, ec3f, ec5f, ec3b, ec3his_t, ec5his_t);
  }
}

// act_cell = ca1 @ wca1act + actbias ; copy final states into d_out
__global__ void __launch_bounds__(256) k_final(
    const float* __restrict__ ec3f, const float* __restrict__ ec5f,
    const float* __restrict__ ca1f, const float* __restrict__ wact,
    const float* __restrict__ abias, float* __restrict__ out) {
  int i = blockIdx.x * blockDim.x + threadIdx.x;
  if (i < NN * NN) {
    out[OFF_EC3 + i] = ec3f[i];
    out[OFF_EC5 + i] = ec5f[i];
    out[OFF_CA1 + i] = ca1f[i];
  }
  if (i < NN * 2) {
    int b = i >> 1, a = i & 1;
    float acc = abias[a];
    for (int c = 0; c < NN; ++c) acc = fmaf(ca1f[b * NN + c], wact[c * 2 + a], acc);
    out[OFF_ACT + i] = acc;
  }
}

// ------------------------- launcher -------------------------

extern "C" void kernel_launch(void* const* d_in, const int* in_sizes, int n_in,
                              void* d_out, int out_size, void* d_ws, size_t ws_size,
                              hipStream_t stream) {
  (void)in_sizes; (void)n_in; (void)out_size; (void)ws_size;
  const uint8_t* cue      = (const uint8_t*)d_in[0];  // bool array, 1 byte/elem
  const float*   ec3_last = (const float*)d_in[1];
  const float*   ec5_last = (const float*)d_in[2];
  // d_in[3] (ca1_last) is dead in the reference: the carry is overwritten
  // before it is ever read.
  const float*   ca1bias  = (const float*)d_in[4];
  const float*   wca3ca1  = (const float*)d_in[5];
  const float*   wec3ca1  = (const float*)d_in[6];
  const float*   wca1ec5  = (const float*)d_in[7];
  const float*   wca1act  = (const float*)d_in[8];
  const float*   actbias  = (const float*)d_in[9];
  float* out = (float*)d_out;

  char* ws = (char*)d_ws;
  float*    ec3f = (float*)(ws + (size_t)0);
  float*    ec5f = (float*)(ws + ((size_t)4  << 20));
  float*    ca1f = (float*)(ws + ((size_t)8  << 20));
  bf16*     ec3b = (bf16*) (ws + ((size_t)12 << 20));
  bf16*     ca1b = (bf16*) (ws + ((size_t)14 << 20));
  bf16*     wt1  = (bf16*) (ws + ((size_t)16 << 20));
  bf16*     wt2  = (bf16*) (ws + ((size_t)18 << 20));
  float*    ca3  = (float*)(ws + ((size_t)20 << 20));
  float*    drv  = (float*)(ws + ((size_t)20 << 20) + ((size_t)512 << 10));
  uint32_t* keys = (uint32_t*)(ws + ((size_t)21 << 20));

  dim3 b256(256);
  k_prep<<<dim3((NN * NN + 255) / 256), b256, 0, stream>>>(
      ec3_last, ec5_last, wec3ca1, wca1ec5, ec3f, ec5f, ec3b, wt1, wt2);
  k_keys<<<1, 128, 0, stream>>>(keys);
  k_ca3<<<dim3((TLEN * NN + 255) / 256), b256, 0, stream>>>(ca3);
  k_drive<<<dim3((TLEN * NN + 255) / 256), b256, 0, stream>>>(ca3, wca3ca1, drv);

  dim3 gemm_grid(8, 16);  // WG tile 64(m) x 128(n); 8 waves of 32x32 tiles
  for (int t = 0; t < TLEN; ++t) {
    k_step1<<<gemm_grid, b256, 0, stream>>>(
        ec3b, wt1, drv + t * NN, ca1bias, ca1f, ca1b, out + OFF_C1H + t * NN);
    k_step2<<<gemm_grid, b256, 0, stream>>>(
        ca1b, wt2, cue, keys, t, ec3f, ec5f, ec3b,
        out + OFF_E3H + t * NN, out + OFF_E5H + t * NN);
  }

  k_final<<<dim3((NN * NN + 255) / 256), b256, 0, stream>>>(
      ec3f, ec5f, ca1f, wca1act, actbias, out);
}
